// Mini_pointgnn_v2_67310727463236
// MI455X (gfx1250) — matrix operand order repaired and checked by reading, not compiled, not run on hardware
//
#include <hip/hip_runtime.h>

typedef __attribute__((ext_vector_type(16))) _Float16 v16h;
typedef __attribute__((ext_vector_type(8)))  _Float16 h8;
typedef __attribute__((ext_vector_type(8)))  float    v8f;

#define Nn   100000
#define Mm   12500
#define E0n  1600000
#define E1n  400000

#define CAT16(lo, hi) __builtin_shufflevector((lo), (hi), 0,1,2,3,4,5,6,7,8,9,10,11,12,13,14,15)

// ---- WMMA fragment fetch (CDNA5 ISA 16-bit layouts) ----
// A 16x32: lane<16 -> M=lane, elems 0..7 = K base+0..7, elems 8..15 = K base+16..23;
//          lanes>=16 shifted by +8 in K.  Two aligned 8-half runs -> 2x ds_load_b128.
__device__ __forceinline__ v16h frag_a(const _Float16* row, int kc, int lane) {
  const h8* r = (const h8*)row;
  int o = kc * 4 + ((lane >= 16) ? 1 : 0);
  return CAT16(r[o], r[o + 2]);
}
// B 32x16 (K-major LDS row per output column): lane<16 -> elems 0..7 = K base+0..7,
// elems 8..15 = K base+16..23; lanes>=16 shifted by +8 in K. Two aligned 8-half runs -> 2x ds_load_b128.
__device__ __forceinline__ v16h frag_b(const _Float16* row, int kc, int lane) {
  const h8* r = (const h8*)row;
  int o = kc * 4 + ((lane >= 16) ? 1 : 0);
  return CAT16(r[o], r[o + 2]);
}

// ======================= simple prologue kernels ========================
__global__ __launch_bounds__(256)
void k_rel(const float* __restrict__ points, const float* __restrict__ centers,
           const int* __restrict__ labels, float* __restrict__ rel) {
  int i = blockIdx.x * 256 + threadIdx.x;
  if (i < Nn) {
    int l = labels[i];
    rel[i*3+0] = points[i*3+0] - centers[l*3+0];
    rel[i*3+1] = points[i*3+1] - centers[l*3+1];
    rel[i*3+2] = points[i*3+2] - centers[l*3+2];
  }
}

__global__ __launch_bounds__(256)
void k_f1(const float* __restrict__ features, const float* __restrict__ rel,
          const float* __restrict__ W, const float* __restrict__ b,
          float* __restrict__ out) {
  int idx = blockIdx.x * 256 + threadIdx.x;            // N*64 threads
  if (idx >= Nn * 64) return;
  int i = idx >> 6, col = idx & 63;
  float acc = b[col];
#pragma unroll
  for (int k = 0; k < 4; ++k) acc += features[i*4+k] * W[k*64+col];
#pragma unroll
  for (int k = 0; k < 3; ++k) acc += rel[i*3+k] * W[(4+k)*64+col];
  out[i*64+col] = fmaxf(acc, 0.0f);
}

__global__ __launch_bounds__(256)
void k_classifier(const float* __restrict__ f, const float* __restrict__ W,
                  const float* __restrict__ b, float* __restrict__ out) {
  int idx = blockIdx.x * 256 + threadIdx.x;            // N*8 threads
  if (idx >= Nn * 8) return;
  int i = idx >> 3, c = idx & 7;
  float acc = b[c];
#pragma unroll 8
  for (int k = 0; k < 64; ++k) acc += f[i*64+k] * W[k*8+c];
  out[i*8+c] = acc;
}

// =================== fused edge MLP + scatter-max (WMMA) ===============
// h = relu([pos[src]-pos[dst], feat[src]] @ W(67x64) + b); agg[dst] = max(agg[dst], h)
// agg pre-zeroed; relu => h>=0 => u32 atomic max on float bits is exact.
__global__ __launch_bounds__(256)
void k_edge_mlp(const float* __restrict__ feat, const float* __restrict__ pos,
                const int* __restrict__ edges, const float* __restrict__ W,
                const float* __restrict__ bias, float* __restrict__ agg,
                int n_edges) {
  __shared__ __align__(16) _Float16 sWt[64 * 96];     // K-major: row n, 96 K (67 padded)
  __shared__ float    sB[64];
  __shared__ __align__(16) _Float16 sA[8][16][96];    // per-wave 16-edge tile
  __shared__ int      sDst[8][16];

  const int tid = threadIdx.x;
  const int w = tid >> 5, lane = tid & 31;

  for (int idx = tid; idx < 96 * 64; idx += 256) {    // coalesced global read, transpose to LDS
    int k = idx >> 6, n = idx & 63;
    sWt[n * 96 + k] = (_Float16)((k < 67) ? W[k*64 + n] : 0.0f);
  }
  if (tid < 64) sB[tid] = bias[tid];

  const int tileBase = blockIdx.x * 128 + w * 16;
  {
    int m = lane >> 1;
    int e = tileBase + m;
    bool valid = (e < n_edges);
    int ec = valid ? e : 0;
    int src = edges[2*ec], dst = edges[2*ec+1];
    if ((lane & 1) == 0) sDst[w][m] = valid ? dst : -1;
    int kbase = (lane & 1) * 48;
#pragma unroll
    for (int g = 0; g < 6; ++g) {                     // 8-wide LDS stores (ds_store_b128)
      int k0 = kbase + g * 8;
      h8 v;
#pragma unroll
      for (int j = 0; j < 8; ++j) {
        int k = k0 + j;
        float f = 0.0f;
        if (k < 3)        f = pos[src*3+k] - pos[dst*3+k];
        else if (k < 67)  f = feat[src*64 + (k-3)];
        v[j] = (_Float16)f;
      }
      *(h8*)&sA[w][m][k0] = v;
    }
  }
  __syncthreads();

  const int mr = lane & 15;
  const _Float16* rowA = &sA[w][mr][0];
  v16h a[3];
#pragma unroll
  for (int kc = 0; kc < 3; ++kc) a[kc] = frag_a(rowA, kc, lane);

#pragma unroll
  for (int nc = 0; nc < 4; ++nc) {
    v8f acc = {};
    const _Float16* rowB = &sWt[(nc*16 + mr) * 96];
#pragma unroll
    for (int kc = 0; kc < 3; ++kc) {
      v16h bb = frag_b(rowB, kc, lane);
      acc = __builtin_amdgcn_wmma_f32_16x16x32_f16(false, a[kc], false, bb,
                                                   (short)0, acc, false, false);
    }
    int col = nc*16 + mr;
    float bv = sB[col];
#pragma unroll
    for (int r = 0; r < 8; ++r) {
      int row = r + ((lane >= 16) ? 8 : 0);
      int dn = sDst[w][row];
      float v = acc[r] + bv;
      if (dn >= 0 && v > 0.0f)
        atomicMax((unsigned int*)&agg[dn*64 + col], __float_as_uint(v));
    }
  }
}

// =================== node update MLP (WMMA, 64x64) =====================
// out = relu(in @ W(64x64) + b) [+ feat] [+ extra]
__global__ __launch_bounds__(256)
void k_node_mlp(const float* __restrict__ in, const float* __restrict__ W,
                const float* __restrict__ bias, const float* __restrict__ feat,
                const float* __restrict__ extra, float* __restrict__ out, int n) {
  __shared__ __align__(16) _Float16 sWt[64 * 64];     // K-major
  __shared__ float    sB[64];
  __shared__ __align__(16) _Float16 sA[8][16][64];

  const int tid = threadIdx.x, w = tid >> 5, lane = tid & 31;
  for (int idx = tid; idx < 64 * 64; idx += 256) {
    int k = idx >> 6, nn = idx & 63;
    sWt[nn * 64 + k] = (_Float16)W[idx];
  }
  if (tid < 64) sB[tid] = bias[tid];

  const int tileBase = blockIdx.x * 128 + w * 16;
  {
    int m = lane >> 1;
    int i = tileBase + m;
    int ic = (i < n) ? i : 0;
    int kbase = (lane & 1) * 32;
    const float4* rin = (const float4*)&in[ic*64 + kbase];  // global_load_b128
#pragma unroll
    for (int g = 0; g < 4; ++g) {
      float4 x = rin[g*2], y = rin[g*2 + 1];
      h8 v;
      v[0] = (_Float16)x.x; v[1] = (_Float16)x.y; v[2] = (_Float16)x.z; v[3] = (_Float16)x.w;
      v[4] = (_Float16)y.x; v[5] = (_Float16)y.y; v[6] = (_Float16)y.z; v[7] = (_Float16)y.w;
      *(h8*)&sA[w][m][kbase + g*8] = v;
    }
  }
  __syncthreads();

  const int mr = lane & 15;
  const _Float16* rowA = &sA[w][mr][0];
  v16h a[2];
#pragma unroll
  for (int kc = 0; kc < 2; ++kc) a[kc] = frag_a(rowA, kc, lane);

#pragma unroll
  for (int nc = 0; nc < 4; ++nc) {
    v8f acc = {};
    const _Float16* rowB = &sWt[(nc*16 + mr) * 64];
#pragma unroll
    for (int kc = 0; kc < 2; ++kc) {
      v16h bb = frag_b(rowB, kc, lane);
      acc = __builtin_amdgcn_wmma_f32_16x16x32_f16(false, a[kc], false, bb,
                                                   (short)0, acc, false, false);
    }
    int col = nc*16 + mr;
    float bv = sB[col];
#pragma unroll
    for (int r = 0; r < 8; ++r) {
      int row = r + ((lane >= 16) ? 8 : 0);
      int i = tileBase + row;
      if (i < n) {
        float v = fmaxf(acc[r] + bv, 0.0f);
        if (feat)  v += feat[i*64+col];
        if (extra) v += extra[i*64+col];
        out[i*64+col] = v;
      }
    }
  }
}

// ========== per-point MLP over [feat(opt gathered by label), rel] =========
// h = relu([src, rel] @ W(67x64) + b); either store, or atomicAdd into csum[labels[i]]
__global__ __launch_bounds__(256)
void k_point_mlp(const float* __restrict__ feat, const float* __restrict__ rel,
                 const int* __restrict__ labels, const float* __restrict__ W,
                 const float* __restrict__ bias, float* __restrict__ out,
                 float* __restrict__ csum, int n, int gatherLabels) {
  __shared__ __align__(16) _Float16 sWt[64 * 96];     // K-major, 67 padded to 96
  __shared__ float    sB[64];
  __shared__ __align__(16) _Float16 sA[8][16][96];

  const int tid = threadIdx.x, w = tid >> 5, lane = tid & 31;
  for (int idx = tid; idx < 96 * 64; idx += 256) {
    int k = idx >> 6, nn = idx & 63;
    sWt[nn * 96 + k] = (_Float16)((k < 67) ? W[k*64 + nn] : 0.0f);
  }
  if (tid < 64) sB[tid] = bias[tid];

  const int tileBase = blockIdx.x * 128 + w * 16;
  {
    int m = lane >> 1;
    int i = tileBase + m;
    int ic = (i < n) ? i : 0;
    int g = gatherLabels ? labels[ic] : ic;
    int kbase = (lane & 1) * 48;
#pragma unroll
    for (int gg = 0; gg < 6; ++gg) {
      int k0 = kbase + gg * 8;
      h8 v;
#pragma unroll
      for (int j = 0; j < 8; ++j) {
        int k = k0 + j;
        float f = 0.0f;
        if (k < 64)       f = feat[g*64 + k];
        else if (k < 67)  f = rel[ic*3 + (k - 64)];
        v[j] = (_Float16)f;
      }
      *(h8*)&sA[w][m][k0] = v;
    }
  }
  __syncthreads();

  const int mr = lane & 15;
  const _Float16* rowA = &sA[w][mr][0];
  v16h a[3];
#pragma unroll
  for (int kc = 0; kc < 3; ++kc) a[kc] = frag_a(rowA, kc, lane);

#pragma unroll
  for (int nc = 0; nc < 4; ++nc) {
    v8f acc = {};
    const _Float16* rowB = &sWt[(nc*16 + mr) * 96];
#pragma unroll
    for (int kc = 0; kc < 3; ++kc) {
      v16h bb = frag_b(rowB, kc, lane);
      acc = __builtin_amdgcn_wmma_f32_16x16x32_f16(false, a[kc], false, bb,
                                                   (short)0, acc, false, false);
    }
    int col = nc*16 + mr;
    float bv = sB[col];
#pragma unroll
    for (int r = 0; r < 8; ++r) {
      int row = r + ((lane >= 16) ? 8 : 0);
      int i = tileBase + row;
      if (i < n) {
        float h = fmaxf(acc[r] + bv, 0.0f);
        if (csum) atomicAdd(&csum[labels[i]*64 + col], h);
        else      out[i*64 + col] = h;
      }
    }
  }
}

// ============================ launcher =================================
extern "C" void kernel_launch(void* const* d_in, const int* in_sizes, int n_in,
                              void* d_out, int out_size, void* d_ws, size_t ws_size,
                              hipStream_t stream) {
  const float* features = (const float*)d_in[0];
  const float* points   = (const float*)d_in[1];
  const float* centers  = (const float*)d_in[2];
  const int*   l0e      = (const int*)d_in[3];
  const int*   l1e      = (const int*)d_in[4];
  const int*   labels   = (const int*)d_in[5];
  const float* W_fe     = (const float*)d_in[6];
  const float* b_fe     = (const float*)d_in[7];
  const float* mini_We  = (const float*)d_in[8];
  const float* mini_be  = (const float*)d_in[9];
  const float* mini_Wu  = (const float*)d_in[10];
  const float* mini_bu  = (const float*)d_in[11];
  const float* W_m1     = (const float*)d_in[12];
  const float* b_m1     = (const float*)d_in[13];
  const float* W_m2     = (const float*)d_in[14];
  const float* b_m2     = (const float*)d_in[15];
  const float* gnn_We   = (const float*)d_in[16];
  const float* gnn_be   = (const float*)d_in[17];
  const float* gnn_Wu   = (const float*)d_in[18];
  const float* gnn_bu   = (const float*)d_in[19];
  const float* W_l      = (const float*)d_in[20];
  const float* b_l      = (const float*)d_in[21];
  const float* W_c      = (const float*)d_in[22];
  const float* b_c      = (const float*)d_in[23];
  float* outp = (float*)d_out;

  float* base = (float*)d_ws;
  float* rel  = base; base += (size_t)Nn*3;
  float* buf0 = base; base += (size_t)Nn*64;   // f1 -> f5 -> final
  float* f2   = base; base += (size_t)Nn*64;
  float* f21  = base; base += (size_t)Nn*64;
  float* f6   = base; base += (size_t)Nn*64;
  float* agg  = base; base += (size_t)Nn*64;
  float* cbuf = base; base += (size_t)Mm*64;
  float* f3   = base; base += (size_t)Mm*64;
  float* f4   = base; base += (size_t)Mm*64;
  float* f41  = base; base += (size_t)Mm*64;
  float* aggM = base; base += (size_t)Mm*64;

  dim3 blk(256);
  auto cdiv = [](int a, int b) { return (a + b - 1) / b; };

  k_rel<<<cdiv(Nn,256), blk, 0, stream>>>(points, centers, labels, rel);
  k_f1<<<cdiv(Nn*64,256), blk, 0, stream>>>(features, rel, W_fe, b_fe, buf0);

  // mini layer 0: f2
  hipMemsetAsync(agg, 0, (size_t)Nn*64*sizeof(float), stream);
  k_edge_mlp<<<cdiv(E0n,128), blk, 0, stream>>>(buf0, points, l0e, mini_We + 0*67*64, mini_be + 0*64, agg, E0n);
  k_node_mlp<<<cdiv(Nn,128), blk, 0, stream>>>(agg, mini_Wu + 0*64*64, mini_bu + 0*64, buf0, nullptr, f2, Nn);
  // mini layer 1: f2_1
  hipMemsetAsync(agg, 0, (size_t)Nn*64*sizeof(float), stream);
  k_edge_mlp<<<cdiv(E0n,128), blk, 0, stream>>>(f2, points, l0e, mini_We + 1*67*64, mini_be + 1*64, agg, E0n);
  k_node_mlp<<<cdiv(Nn,128), blk, 0, stream>>>(agg, mini_Wu + 1*64*64, mini_bu + 1*64, f2, nullptr, f21, Nn);

  // pooling: c = segment_sum(relu([f2_1, rel] @ W_m1 + b_m1)); f3 = relu(c @ W_m2 + b_m2)
  hipMemsetAsync(cbuf, 0, (size_t)Mm*64*sizeof(float), stream);
  k_point_mlp<<<cdiv(Nn,128), blk, 0, stream>>>(f21, rel, labels, W_m1, b_m1, nullptr, cbuf, Nn, 0);
  k_node_mlp<<<cdiv(Mm,128), blk, 0, stream>>>(cbuf, W_m2, b_m2, nullptr, nullptr, f3, Mm);

  // cluster GNN layer 0: f4
  hipMemsetAsync(aggM, 0, (size_t)Mm*64*sizeof(float), stream);
  k_edge_mlp<<<cdiv(E1n,128), blk, 0, stream>>>(f3, centers, l1e, gnn_We + 0*67*64, gnn_be + 0*64, aggM, E1n);
  k_node_mlp<<<cdiv(Mm,128), blk, 0, stream>>>(aggM, gnn_Wu + 0*64*64, gnn_bu + 0*64, f3, nullptr, f4, Mm);
  // cluster GNN layer 1: f4_1
  hipMemsetAsync(aggM, 0, (size_t)Mm*64*sizeof(float), stream);
  k_edge_mlp<<<cdiv(E1n,128), blk, 0, stream>>>(f4, centers, l1e, gnn_We + 1*67*64, gnn_be + 1*64, aggM, E1n);
  k_node_mlp<<<cdiv(Mm,128), blk, 0, stream>>>(aggM, gnn_Wu + 1*64*64, gnn_bu + 1*64, f4, nullptr, f41, Mm);

  // f5 = relu([f4_1[labels], rel] @ W_l + b_l)  -> buf0
  k_point_mlp<<<cdiv(Nn,128), blk, 0, stream>>>(f41, rel, labels, W_l, b_l, buf0, nullptr, Nn, 1);

  // mini layer 2: f6 = relu(agg@Wu+bu) + f5 + f2_1
  hipMemsetAsync(agg, 0, (size_t)Nn*64*sizeof(float), stream);
  k_edge_mlp<<<cdiv(E0n,128), blk, 0, stream>>>(buf0, points, l0e, mini_We + 2*67*64, mini_be + 2*64, agg, E0n);
  k_node_mlp<<<cdiv(Nn,128), blk, 0, stream>>>(agg, mini_Wu + 2*64*64, mini_bu + 2*64, buf0, f21, f6, Nn);

  // mini layer 3: final = relu(agg@Wu+bu) + f6 + f2  -> buf0
  hipMemsetAsync(agg, 0, (size_t)Nn*64*sizeof(float), stream);
  k_edge_mlp<<<cdiv(E0n,128), blk, 0, stream>>>(f6, points, l0e, mini_We + 3*67*64, mini_be + 3*64, agg, E0n);
  k_node_mlp<<<cdiv(Nn,128), blk, 0, stream>>>(agg, mini_Wu + 3*64*64, mini_bu + 3*64, f6, f2, buf0, Nn);

  // classifier head
  k_classifier<<<cdiv(Nn*8,256), blk, 0, stream>>>(buf0, W_c, b_c, outp);
}
